// SphereNet_23141283791012
// MI455X (gfx1250) — compile-verified, matrix operand order checked by reading, not run on hardware
//
#include <hip/hip_runtime.h>
#include <hip/hip_bf16.h>

// ============================================================================
// SphereNet-style GNN for MI455X (gfx1250, wave32, WMMA).
//
// Key optimizations:
//  * All dense layers run through a WMMA f32_16x16x32_f16 GEMM with fused
//    bias/swish/residual/accumulate/elementwise-multiply epilogue.
//  * Weight slice (64 cols x kpad, <= 40KB) staged per block into LDS
//    (b128 copies + one barrier); B fragments come from ds_load_b128 while A
//    streams from global -- DScnt and LOADcnt overlap.
//  * Two A formats: fp32 (K % 32 == 0, converted in-register) and fp16
//    (pre-padded).  sbf/t are layer-invariant -> padded f16 copies made ONCE,
//    halving the biggest HBM stream and removing all K-tail guard code
//    (the old guarded path compiled to divergent per-element masked loads).
//  * sb*tt fused via epilogue multiply (one T*64 buffer and stream removed);
//    x_kj * rbfE fused into the lin_kj GEMM epilogue.
//  * seg_mean(x[idx], idx, n) == (touched ? x : 0)  -> flag-select kernels.
//  * Only the final _update_v is computed (earlier v's are dead).
//  * rank-8 basis projections are pre-composed: (X@W1^T)@W2^T == X@(W2@W1)^T.
//  * E=300000, T=400000, N=20000 are all multiples of 16 -> no M tails.
//
// Workspace requirement: ~1.4 GB.
// ============================================================================

typedef __attribute__((ext_vector_type(16))) _Float16 v16h;
typedef __attribute__((ext_vector_type(8)))  _Float16 v8h;
typedef __attribute__((ext_vector_type(8)))  float    v8f;
typedef __attribute__((ext_vector_type(4)))  float    v4f;

__device__ __forceinline__ float swishf(float x) {
  return x * (1.0f / (1.0f + __expf(-x)));
}

// ----------------------------------------------------------------------------
// WMMA GEMM:  Y[M,Nout] = epilogue( X[M,K] @ Wt^T ) with Wt f16 [Nout, kpad].
// A source: Xf (fp32, ldx=K, K%32==0) or Xh (fp16, ldx=A kpad).  Optional row
// gather on A, addto (Y += before bias), bias, swish, residual add after
// activation, elementwise multiply last.  Each wave owns a 16x64 tile; the
// block's 64-column weight slice lives in LDS.
// ----------------------------------------------------------------------------
#define KPAD_MAX 320

__global__ __launch_bounds__(256)
void wmma_gemm_kernel(const float* __restrict__ Xf,
                      const _Float16* __restrict__ Xh, int ldx,
                      const int* __restrict__ rowidx,
                      const _Float16* __restrict__ Wt, int kpad, int K,
                      const float* __restrict__ bias,
                      const float* __restrict__ resid,
                      const float* __restrict__ mulsrc,
                      float* __restrict__ Y, int ldy,
                      int M, int act, int addto)
{
  __shared__ __align__(16) _Float16 Wlds[64 * KPAD_MAX];   // 40 KB max

  const int tid  = threadIdx.x;
  const int col0 = blockIdx.y * 64;

  // Cooperative stage of the contiguous weight slice rows [col0, col0+64).
  {
    const uint4* src = (const uint4*)(Wt + (size_t)col0 * (size_t)kpad);
    uint4*       dst = (uint4*)Wlds;
    const int nvec = (64 * kpad) >> 3;              // halves / 8 per uint4
    for (int i = tid; i < nvec; i += 256) dst[i] = src[i];
  }
  __syncthreads();                                  // all threads reach this

  const int lane = tid & 31;
  const int wave = tid >> 5;
  const int row0 = (blockIdx.x * 8 + wave) * 16;
  if (row0 >= M) return;                 // wave-uniform: EXEC all-1s for WMMA

  const int m   = lane & 15;             // A row / B col / C col within tile
  const int hi  = lane >> 4;             // half-wave select (ISA 16-bit layout)
  const int kb0 = hi ? 8  : 0;
  const int kb1 = hi ? 24 : 16;

  int arow = row0 + m;
  if (rowidx) arow = rowidx[arow];

  v8f acc[4] = {};

  if (Xh == nullptr) {
    // ---------------- fp32 A path (K % 32 == 0), software-pipelined --------
    const float* Arow = Xf + (size_t)arow * (size_t)ldx;
    v4f cx0 = ((const v4f*)(Arow + kb0))[0];
    v4f cx1 = ((const v4f*)(Arow + kb0))[1];
    v4f cy0 = ((const v4f*)(Arow + kb1))[0];
    v4f cy1 = ((const v4f*)(Arow + kb1))[1];
    for (int kk = 0; kk < K; kk += 32) {
      v16h a;
      #pragma unroll
      for (int t = 0; t < 4; ++t) {
        a[t]      = (_Float16)cx0[t];
        a[4 + t]  = (_Float16)cx1[t];
        a[8 + t]  = (_Float16)cy0[t];
        a[12 + t] = (_Float16)cy1[t];
      }
      if (kk + 32 < K) {                 // next-iter loads fly during WMMAs
        const int kn = kk + 32;
        cx0 = ((const v4f*)(Arow + kn + kb0))[0];
        cx1 = ((const v4f*)(Arow + kn + kb0))[1];
        cy0 = ((const v4f*)(Arow + kn + kb1))[0];
        cy1 = ((const v4f*)(Arow + kn + kb1))[1];
      }
      v8h b0[4], b1[4];
      #pragma unroll
      for (int nt = 0; nt < 4; ++nt) {
        const _Float16* Wr = Wlds + (nt * 16 + m) * kpad + kk;
        b0[nt] = *(const v8h*)(Wr + kb0);
        b1[nt] = *(const v8h*)(Wr + kb1);
      }
      #pragma unroll
      for (int nt = 0; nt < 4; ++nt) {
        v16h b;
        #pragma unroll
        for (int t = 0; t < 8; ++t) { b[t] = b0[nt][t]; b[8 + t] = b1[nt][t]; }
        acc[nt] = __builtin_amdgcn_wmma_f32_16x16x32_f16(
            false, a, false, b, (short)0, acc[nt], false, false);
      }
    }
  } else {
    // ---------------- fp16 A path (pre-padded, ldx = A kpad) ---------------
    const _Float16* ArowH = Xh + (size_t)arow * (size_t)ldx;
    v8h hx0 = *(const v8h*)(ArowH + kb0);
    v8h hx1 = *(const v8h*)(ArowH + kb1);
    for (int kk = 0; kk < K; kk += 32) {
      v16h a;
      #pragma unroll
      for (int t = 0; t < 8; ++t) { a[t] = hx0[t]; a[8 + t] = hx1[t]; }
      if (kk + 32 < K) {
        hx0 = *(const v8h*)(ArowH + kk + 32 + kb0);
        hx1 = *(const v8h*)(ArowH + kk + 32 + kb1);
      }
      v8h b0[4], b1[4];
      #pragma unroll
      for (int nt = 0; nt < 4; ++nt) {
        const _Float16* Wr = Wlds + (nt * 16 + m) * kpad + kk;
        b0[nt] = *(const v8h*)(Wr + kb0);
        b1[nt] = *(const v8h*)(Wr + kb1);
      }
      #pragma unroll
      for (int nt = 0; nt < 4; ++nt) {
        v16h b;
        #pragma unroll
        for (int t = 0; t < 8; ++t) { b[t] = b0[nt][t]; b[8 + t] = b1[nt][t]; }
        acc[nt] = __builtin_amdgcn_wmma_f32_16x16x32_f16(
            false, a, false, b, (short)0, acc[nt], false, false);
      }
    }
  }

  // C/D layout: VGPR r -> row (row0 + r + 8*hi), col = col0 + nt*16 + (lane&15)
  const int mo = hi ? 8 : 0;
  #pragma unroll
  for (int nt = 0; nt < 4; ++nt) {
    const int col = col0 + nt * 16 + m;
    const float bv = bias ? bias[col] : 0.0f;
    #pragma unroll
    for (int r = 0; r < 8; ++r) {
      size_t o = (size_t)(row0 + mo + r) * (size_t)ldy + col;
      float zz = acc[nt][r];
      if (addto)  zz += Y[o];
      zz += bv;
      if (act)    zz = swishf(zz);
      if (resid)  zz += resid[o];
      if (mulsrc) zz *= mulsrc[o];
      Y[o] = zz;
    }
  }
}

// ---------------- small / irregular kernels ---------------------------------

__global__ void small_linear_kernel(const float* __restrict__ X, int K,
                                    const float* __restrict__ W,
                                    const float* __restrict__ bias,
                                    const float* __restrict__ mul,
                                    float* __restrict__ Y,
                                    int Nout, int M, int act)
{
  int idx = blockIdx.x * blockDim.x + threadIdx.x;
  if (idx >= M * Nout) return;
  int mr = idx / Nout, n = idx - mr * Nout;
  const float* xr = X + (size_t)mr * K;
  const float* wr = W + (size_t)n * K;
  float s = bias ? bias[n] : 0.0f;
  for (int k = 0; k < K; ++k) s += xr[k] * wr[k];
  if (act) s = swishf(s);
  if (mul) s *= mul[idx];
  Y[idx] = s;
}

// Wc[Nout,K] = W2[Nout,J] @ W1[J,K]
__global__ void compose_kernel(const float* __restrict__ W2,
                               const float* __restrict__ W1,
                               float* __restrict__ Wc, int Nout, int J, int K)
{
  int idx = blockIdx.x * blockDim.x + threadIdx.x;
  if (idx >= Nout * K) return;
  int n = idx / K, k = idx - n * K;
  float s = 0.0f;
  for (int j = 0; j < J; ++j) s += W2[n * J + j] * W1[j * K + k];
  Wc[idx] = s;
}

// dst f16 [Nout, kpad] = f32 src[n*srcld + coloff + k], zero-padded k>=K
__global__ void cvt_weight_kernel(const float* __restrict__ src, int srcld,
                                  int coloff, _Float16* __restrict__ dst,
                                  int K, int kpad, int Nout)
{
  int idx = blockIdx.x * blockDim.x + threadIdx.x;
  if (idx >= Nout * kpad) return;
  int n = idx / kpad, k = idx - n * kpad;
  dst[idx] = (k < K) ? (_Float16)src[(size_t)n * srcld + coloff + k]
                     : (_Float16)0.0f;
}

// dst f16 [M, kpad] = f32 src [M, K], zero-padded k>=K  (activation padding)
__global__ void pad16_kernel(const float* __restrict__ src, int K,
                             _Float16* __restrict__ dst, int kpad, long M)
{
  long idx = (long)blockIdx.x * blockDim.x + threadIdx.x;
  if (idx >= M * kpad) return;
  long mr = idx / kpad;
  int  k  = (int)(idx - mr * kpad);
  dst[idx] = (k < K) ? (_Float16)src[mr * K + k] : (_Float16)0.0f;
}

__global__ void embed_kernel(const int* __restrict__ z,
                             const float* __restrict__ emb,
                             float* __restrict__ x, int N)
{
  int idx = blockIdx.x * blockDim.x + threadIdx.x;
  if (idx >= N * 128) return;
  int n = idx >> 7, h = idx & 127;
  x[idx] = emb[(size_t)z[n] * 128 + h];
}

__global__ void fill_f32_kernel(float* __restrict__ p, float v, long n)
{
  long i = (long)blockIdx.x * blockDim.x + threadIdx.x;
  if (i < n) p[i] = v;
}

__global__ void fill_i32_kernel(int* __restrict__ p, int v, long n)
{
  long i = (long)blockIdx.x * blockDim.x + threadIdx.x;
  if (i < n) p[i] = v;
}

__global__ void mark_edge_kernel(const int* __restrict__ ji,
                                 const int* __restrict__ kj,
                                 int* __restrict__ flag, int T)
{
  int t = blockIdx.x * blockDim.x + threadIdx.x;
  if (t < T) { flag[ji[t]] = 1; flag[kj[t]] = 1; }
}

__global__ void mark_node_kernel(const int* __restrict__ ii,
                                 int* __restrict__ flag, int E)
{
  int e = blockIdx.x * blockDim.x + threadIdx.x;
  if (e < E) flag[ii[e]] = 1;
}

// dst = flag[row] ? nv : dst   (dst already holds the "else" value)
__global__ void select_edge_kernel(const int* __restrict__ flag,
                                   const float* __restrict__ nv,
                                   float* __restrict__ dst, int E)
{
  int idx = blockIdx.x * blockDim.x + threadIdx.x;
  if (idx >= E * 128) return;
  if (flag[idx >> 7]) dst[idx] = nv[idx];
}

// acc[idx_ji[t]] += xk[idx_kj[t]] * st[t]   over [T,64]  (st = sb*tt fused)
__global__ void triplet_kernel(const float* __restrict__ xk,
                               const float* __restrict__ st,
                               const int* __restrict__ kj,
                               const int* __restrict__ ji,
                               float* __restrict__ acc, int T)
{
  long idx = (long)blockIdx.x * blockDim.x + threadIdx.x;
  if (idx >= (long)T * 64) return;
  int t = (int)(idx >> 6), c = (int)(idx & 63);
  float v = xk[(size_t)kj[t] * 64 + c] * st[idx];
  atomicAdd(acc + (size_t)ji[t] * 64 + c, v);
}

__global__ void scatter_node_kernel(const float* __restrict__ e2,
                                    const int* __restrict__ ii,
                                    float* __restrict__ pool, int E)
{
  int idx = blockIdx.x * blockDim.x + threadIdx.x;
  if (idx >= E * 128) return;
  int e = idx >> 7, c = idx & 127;
  atomicAdd(pool + (size_t)ii[e] * 128 + c, e2[idx]);
}

__global__ void final_out_kernel(const int* __restrict__ flag,
                                 const float* __restrict__ v,
                                 float* __restrict__ out, int N)
{
  int idx = blockIdx.x * blockDim.x + threadIdx.x;
  if (idx >= N * 128) return;
  out[idx] = flag[idx >> 7] ? v[idx] : 0.0f;
}

// ============================================================================
// Host orchestration
// ============================================================================
extern "C" void kernel_launch(void* const* d_in, const int* in_sizes, int n_in,
                              void* d_out, int out_size, void* d_ws,
                              size_t ws_size, hipStream_t stream)
{
  (void)in_sizes; (void)n_in; (void)out_size; (void)ws_size;
  const int N = 20000, E = 300000, T = 400000;

  const int*   z   = (const int*)  d_in[0];
  const float* rbf = (const float*)d_in[1];
  const float* sbf = (const float*)d_in[2];
  const float* tsp = (const float*)d_in[3];
  const int*   gi  = (const int*)  d_in[4];
  const int*   gj  = (const int*)  d_in[5];
  const int*   ikj = (const int*)  d_in[6];
  const int*   iji = (const int*)  d_in[7];
  // params leaves in JAX sorted-dict tree-flatten order:
  // init:  0 emb, 1 lin_b, 2 lin_rbf_0_b, 3 lin_rbf_0_w, 4 lin_rbf_1_w, 5 lin_w
  // init_v: 6..13 (unused: its update_v output is dead)
  // layers[L]: e-block at 14+L*35 (+0..7 after, +8..11 before, +12 lin_b,
  //   +13 lin_down, +14 lin_ji_b, +15 lin_ji_w, +16 lin_kj_b, +17 lin_kj_w,
  //   +18 lin_rbf, +19 lin_rbf1, +20 lin_rbf2, +21 lin_sbf1, +22 lin_sbf2,
  //   +23 lin_t1, +24 lin_t2, +25 lin_up, +26 lin_w), v-block at +27
  //   (+0 lin_b, +1 lin_up_b, +2 lin_up_w, +3 lin_w, +4/5 lins0 W/b, +6/7 lins1)
  void* const* P = d_in + 8;
  auto pf = [&](int k) -> const float* { return (const float*)P[k]; };

  size_t off = 0;
  auto carve = [&](size_t bytes) -> void* {
    void* p = (char*)d_ws + off;
    off += (bytes + 255) & ~(size_t)255;
    return p;
  };
  float*    e1   = (float*)carve((size_t)E * 128 * 4);
  float*    e2   = (float*)carve((size_t)E * 128 * 4);
  float*    tA   = (float*)carve((size_t)E * 128 * 4);
  float*    tB   = (float*)carve((size_t)E * 128 * 4);
  float*    tC   = (float*)carve((size_t)E * 128 * 4);
  float*    tt64 = (float*)carve((size_t)T * 64 * 4);   // tt, then sb*tt
  float*    ac64 = (float*)carve((size_t)E * 64 * 4);
  float*    xk64 = (float*)carve((size_t)E * 64 * 4);
  float*    xemb = (float*)carve((size_t)N * 128 * 4);
  float*    pool = (float*)carve((size_t)N * 128 * 4);
  float*    vb1  = (float*)carve((size_t)N * 256 * 4);
  float*    vb2  = (float*)carve((size_t)N * 256 * 4);
  int*      efl  = (int*)  carve((size_t)E * 4);
  int*      nfl  = (int*)  carve((size_t)N * 4);
  float*    cw   = (float*)carve((size_t)64 * 294 * 4);     // composed weights
  _Float16* Wh   = (_Float16*)carve((size_t)256 * 320 * 2); // shared f16 slot
  _Float16* sh16 = (_Float16*)carve((size_t)T * 64 * 2);    // sbf padded f16
  _Float16* th16 = (_Float16*)carve((size_t)T * 320 * 2);   // t   padded f16

  auto gemm = [&](const float* Xf, const _Float16* Xh, int ldx,
                  const int* ridx, int kpad, int K,
                  const float* bias, const float* resid, const float* mulsrc,
                  float* Y, int Nout, int M, int act, int addto) {
    dim3 g((unsigned)((M + 127) / 128), (unsigned)(Nout / 64));
    wmma_gemm_kernel<<<g, dim3(256), 0, stream>>>(Xf, Xh, ldx, ridx, Wh, kpad,
                                                  K, bias, resid, mulsrc, Y,
                                                  Nout, M, act, addto);
  };
  auto cvtw = [&](const float* src, int srcld, int coloff, int Nout, int K,
                  int kpad) {
    int n = Nout * kpad;
    cvt_weight_kernel<<<(n + 255) / 256, 256, 0, stream>>>(src, srcld, coloff,
                                                           Wh, K, kpad, Nout);
  };
  auto slin = [&](const float* X, int K, const float* W, const float* bias,
                  const float* mul, float* Y, int Nout, int M, int act) {
    long n = (long)M * Nout;
    small_linear_kernel<<<(unsigned)((n + 255) / 256), 256, 0, stream>>>(
        X, K, W, bias, mul, Y, Nout, M, act);
  };
  auto zero_f = [&](float* p, long n) {
    fill_f32_kernel<<<(unsigned)((n + 255) / 256), 256, 0, stream>>>(p, 0.0f, n);
  };
  auto zero_i = [&](int* p, long n) {
    fill_i32_kernel<<<(unsigned)((n + 255) / 256), 256, 0, stream>>>(p, 0, n);
  };

  // --------------------------- init stage -----------------------------------
  embed_kernel<<<(N * 128 + 255) / 256, 256, 0, stream>>>(z, pf(0), xemb, N);
  // pad sbf[T,42]->f16[T,64], t[T,294]->f16[T,320] ONCE (layer-invariant)
  pad16_kernel<<<(unsigned)(((long)T * 64 + 255) / 256), 256, 0, stream>>>(
      sbf, 42, sh16, 64, (long)T);
  pad16_kernel<<<(unsigned)(((long)T * 320 + 255) / 256), 256, 0, stream>>>(
      tsp, 294, th16, 320, (long)T);
  // rbf0a = swish(rbf @ lin_rbf_0_w^T + b) -> tC  [E,128]
  slin(rbf, 6, pf(3), pf(2), nullptr, tC, 128, E, 1);
  // e1 = swish([x[i] | x[j] | rbf0a] @ lin_w^T + b) via three K=128 slices
  cvtw(pf(5), 384, 0, 128, 128, 128);
  gemm(xemb, nullptr, 128, gi, 128, 128, nullptr, nullptr, nullptr,
       e1, 128, E, 0, 0);
  cvtw(pf(5), 384, 128, 128, 128, 128);
  gemm(xemb, nullptr, 128, gj, 128, 128, nullptr, nullptr, nullptr,
       e1, 128, E, 0, 1);
  cvtw(pf(5), 384, 256, 128, 128, 128);
  gemm(tC, nullptr, 128, nullptr, 128, 128, pf(1), nullptr, nullptr,
       e1, 128, E, 1, 1);
  // e2 = (rbf @ lin_rbf_1_w^T) * e1
  slin(rbf, 6, pf(4), nullptr, e1, e2, 128, E, 0);
  // init _update_v is dead (v overwritten each layer) -> skipped.

  // --------------------------- layers ---------------------------------------
  for (int L = 0; L < 2; ++L) {
    const int eb = 14 + L * 35;
    auto pe = [&](int k) -> const float* { return (const float*)P[eb + k]; };
    auto pv = [&](int k) -> const float* { return (const float*)P[eb + 27 + k]; };

    // rbfE = rbf @ (lin_rbf2 @ lin_rbf1)^T -> tC
    compose_kernel<<<(128 * 6 + 255) / 256, 256, 0, stream>>>(pe(20), pe(19),
                                                              cw, 128, 8, 6);
    slin(rbf, 6, cw, nullptr, nullptr, tC, 128, E, 0);
    // x_ji = swish(e1 @ lin_ji^T + b) -> tA
    cvtw(pe(15), 128, 0, 128, 128, 128);
    gemm(e1, nullptr, 128, nullptr, 128, 128, pe(14), nullptr, nullptr,
         tA, 128, E, 1, 0);
    // x_kj = swish(e1 @ lin_kj^T + b) * rbfE -> tB   (multiply fused)
    cvtw(pe(17), 128, 0, 128, 128, 128);
    gemm(e1, nullptr, 128, nullptr, 128, 128, pe(16), nullptr, tC,
         tB, 128, E, 1, 0);
    // xk64 = swish(x_kj @ lin_down^T)  [E,64]
    cvtw(pe(13), 128, 0, 64, 128, 128);
    gemm(tB, nullptr, 128, nullptr, 128, 128, nullptr, nullptr, nullptr,
         xk64, 64, E, 1, 0);
    // tt64 = t @ (lin_t2 @ lin_t1)^T  [T,64]   (f16 A, K=320 padded)
    compose_kernel<<<(64 * 294 + 255) / 256, 256, 0, stream>>>(pe(24), pe(23),
                                                               cw, 64, 8, 294);
    cvtw(cw, 294, 0, 64, 294, 320);
    gemm(nullptr, th16, 320, nullptr, 320, 320, nullptr, nullptr, nullptr,
         tt64, 64, T, 0, 0);
    // tt64 = (sbf @ (lin_sbf2 @ lin_sbf1)^T) * tt64  (sb*tt fused, in-place)
    compose_kernel<<<(64 * 42 + 255) / 256, 256, 0, stream>>>(pe(22), pe(21),
                                                              cw, 64, 8, 42);
    cvtw(cw, 42, 0, 64, 42, 64);
    gemm(nullptr, sh16, 64, nullptr, 64, 64, nullptr, nullptr, tt64,
         tt64, 64, T, 0, 0);
    // acc64 = segment_sum(xk64[idx_kj] * (sb*tt), idx_ji)
    zero_f(ac64, (long)E * 64);
    triplet_kernel<<<(unsigned)(((long)T * 64 + 255) / 256), 256, 0, stream>>>(
        xk64, tt64, ikj, iji, ac64, T);
    // e1b = x_ji + swish(acc64 @ lin_up^T) -> tB
    cvtw(pe(25), 64, 0, 128, 64, 64);
    gemm(ac64, nullptr, 64, nullptr, 64, 64, nullptr, tA, nullptr,
         tB, 128, E, 1, 0);
    // before block (NB=1):  tB += swish(swish(tB@w1+b1)@w2+b2)
    cvtw(pe(8), 128, 0, 128, 128, 128);
    gemm(tB, nullptr, 128, nullptr, 128, 128, pe(9), nullptr, nullptr,
         tC, 128, E, 1, 0);
    cvtw(pe(10), 128, 0, 128, 128, 128);
    gemm(tC, nullptr, 128, nullptr, 128, 128, pe(11), tB, nullptr,
         tB, 128, E, 1, 0);
    // tB = swish(tB @ lin_w + b) + x1(e1)
    cvtw(pe(26), 128, 0, 128, 128, 128);
    gemm(tB, nullptr, 128, nullptr, 128, 128, pe(12), e1, nullptr,
         tB, 128, E, 1, 0);
    // after blocks (NA=2)
    for (int ab = 0; ab < 2; ++ab) {
      cvtw(pe(ab * 4 + 0), 128, 0, 128, 128, 128);
      gemm(tB, nullptr, 128, nullptr, 128, 128, pe(ab * 4 + 1), nullptr,
           nullptr, tC, 128, E, 1, 0);
      cvtw(pe(ab * 4 + 2), 128, 0, 128, 128, 128);
      gemm(tC, nullptr, 128, nullptr, 128, 128, pe(ab * 4 + 3), tB, nullptr,
           tB, 128, E, 1, 0);
    }
    // e2 candidate = (rbf @ lin_rbf^T) * e1b -> tC
    slin(rbf, 6, pe(18), nullptr, tB, tC, 128, E, 0);
    // seg_mean over gathered identical values == flag-select
    zero_i(efl, (long)E);
    mark_edge_kernel<<<(T + 255) / 256, 256, 0, stream>>>(iji, ikj, efl, T);
    select_edge_kernel<<<(unsigned)(((long)E * 128 + 255) / 256), 256, 0,
                         stream>>>(efl, tB, e1, E);
    select_edge_kernel<<<(unsigned)(((long)E * 128 + 255) / 256), 256, 0,
                         stream>>>(efl, tC, e2, E);

    // only the final layer's update_v is live
    if (L == 1) {
      zero_f(pool, (long)N * 128);
      scatter_node_kernel<<<(unsigned)(((long)E * 128 + 255) / 256), 256, 0,
                            stream>>>(e2, gi, pool, E);
      cvtw(pv(2), 128, 0, 256, 128, 128);      // lin_up_w [256,128]
      gemm(pool, nullptr, 128, nullptr, 128, 128, pv(1), nullptr, nullptr,
           vb1, 256, N, 0, 0);
      cvtw(pv(4), 256, 0, 256, 256, 256);      // lins[0].W
      gemm(vb1, nullptr, 256, nullptr, 256, 256, pv(5), nullptr, nullptr,
           vb2, 256, N, 1, 0);
      cvtw(pv(6), 256, 0, 256, 256, 256);      // lins[1].W
      gemm(vb2, nullptr, 256, nullptr, 256, 256, pv(7), nullptr, nullptr,
           vb1, 256, N, 1, 0);
      cvtw(pv(3), 256, 0, 128, 256, 256);      // lin_w [128,256]
      gemm(vb1, nullptr, 256, nullptr, 256, 256, pv(0), nullptr, nullptr,
           vb2, 128, N, 0, 0);
      zero_i(nfl, (long)N);
      mark_node_kernel<<<(E + 255) / 256, 256, 0, stream>>>(gi, nfl, E);
      final_out_kernel<<<(N * 128 + 255) / 256, 256, 0, stream>>>(
          nfl, vb2, (float*)d_out, N);
    }
  }
}
